// ContextKGEModel_58729382806062
// MI455X (gfx1250) — matrix operand order repaired and checked by reading, not compiled, not run on hardware
//
#include <hip/hip_runtime.h>
#include <hip/hip_bf16.h>

// ---------------------------------------------------------------------------
// Problem constants (from reference): B=128, NEG=15 -> G=2048 groups,
// L=32 max slots per group, D=768, hidden = 4*D = 3072.
// ---------------------------------------------------------------------------
#define BATCH   128
#define NEGS    15
#define GRP     2048          // BATCH * (NEGS + 1)
#define LMAX    32
#define DIM     768
#define DHID    3072          // 4 * DIM
#define ECHUNK  128           // e-dimension chunk for XW to fit LDS
#define GAMMA_F 0.1f

typedef __attribute__((ext_vector_type(16))) __bf16        v16bf;
typedef __attribute__((ext_vector_type(8)))  float         v8f;
typedef __attribute__((ext_vector_type(4)))  unsigned int  u32x4;
typedef __attribute__((ext_vector_type(8)))  unsigned int  u32x8;

// ---------------------------------------------------------------------------
// WMMA fragment helpers (wave32, v_wmma_f32_16x16x32_bf16).
// Layouts used here (contiguous K order, same for A and B):
//   A (16x32, MxK):  lane&15 = M row; lane>>4 selects K-half.
//       element e (0..15): K = (lane>>4)*16 + e
//   B (32x16, KxN):  lane&15 = N col; lane>>4 selects K-half.
//       element e (0..15): K = (lane>>4)*16 + e
//   C/D (16x16 f32): lane&15 = N col; VGPR r -> M = r + (lane>>4)*8
// ---------------------------------------------------------------------------
__device__ inline v16bf load_A_bf16(const __bf16* X, int ld, int mbase, int kbase, int lane) {
    int row   = mbase + (lane & 15);
    int khalf = (lane >> 4) * 16;
    v16bf a;
#pragma unroll
    for (int e = 0; e < 16; ++e) {
        int k = kbase + khalf + e;
        a[e] = X[(size_t)row * ld + k];
    }
    return a;
}

// B[k][n] = Src[n][k]  (i.e. multiply by Src^T), Src is bf16 row-major
__device__ inline v16bf load_Bt_bf16(const __bf16* Src, int ld, int nbase, int kbase, int lane) {
    int n  = nbase + (lane & 15);
    int k0 = kbase + (lane >> 4) * 16;
    v16bf b;
#pragma unroll
    for (int e = 0; e < 16; ++e) b[e] = Src[(size_t)n * ld + k0 + e];
    return b;
}

// B[k][n] = W[n][k]  with W f32 row-major (weights streamed from global, cvt to bf16)
__device__ inline v16bf load_Bt_f32(const float* W, int ld, int nbase, int kbase, int lane) {
    int n  = nbase + (lane & 15);
    int k0 = kbase + (lane >> 4) * 16;
    v16bf b;
#pragma unroll
    for (int e = 0; e < 16; ++e) b[e] = (__bf16)W[(size_t)n * ld + k0 + e];
    return b;
}

__device__ inline void store_D_bf16(__bf16* Out, int ld, int mbase, int nbase, int lane, v8f d) {
    int n  = nbase + (lane & 15);
    int m0 = mbase + (lane >> 4) * 8;
#pragma unroll
    for (int r = 0; r < 8; ++r) Out[(size_t)(m0 + r) * ld + n] = (__bf16)d[r];
}

__device__ inline void store_D_f32(float* Out, int ld, int mbase, int nbase, int lane, v8f d) {
    int n  = nbase + (lane & 15);
    int m0 = mbase + (lane >> 4) * 8;
#pragma unroll
    for (int r = 0; r < 8; ++r) Out[(size_t)(m0 + r) * ld + n] = d[r];
}

__device__ inline v8f wmma_bf16(v16bf a, v16bf b, v8f c) {
    return __builtin_amdgcn_wmma_f32_16x16x32_bf16(false, a, false, b, (short)0, c, false, false);
}

// ---------------------------------------------------------------------------
// Kernel 0: exclusive prefix sum of ragged sizes + zero score accumulator
// ---------------------------------------------------------------------------
__global__ __launch_bounds__(256) void setup_kernel(const int* __restrict__ sizes,
                                                    int* __restrict__ offsets,
                                                    float* __restrict__ scoreAcc) {
    int tid = threadIdx.x;
    for (int i = tid; i < GRP; i += 256) scoreAcc[i] = 0.0f;
    if (tid == 0) {
        int acc = 0;
        for (int g = 0; g < GRP; ++g) { offsets[g] = acc; acc += sizes[g]; }
    }
}

// ---------------------------------------------------------------------------
// Kernel 1: per-group bilinear attention + pooling.
// One workgroup (8 waves) per group.
//   Xs  : group rows, f32 -> bf16                (32 x 768, 48 KB)
//   XWs : chunk of X @ W_sfa^T                   (32 x 128,  8 KB)
//   Ss  : Gram matrix X @ XW^T (accum. in VGPRs) (32 x 32,   4 KB)
// ---------------------------------------------------------------------------
__global__ __launch_bounds__(256) void group_kernel(const float* __restrict__ emb,
                                                    const float* __restrict__ Wsfa,
                                                    const int*   __restrict__ offsets,
                                                    const int*   __restrict__ sizes,
                                                    __bf16*      __restrict__ pooled_out) {
    __shared__ __bf16 Xs[LMAX * DIM];
    __shared__ __bf16 XWs[LMAX * ECHUNK];
    __shared__ float  Ss[LMAX * LMAX];
    __shared__ float  att[LMAX];
    __shared__ float  invsum;

    const int g   = blockIdx.x;
    const int sz  = sizes[g];
    const int off = offsets[g];
    const int tid = threadIdx.x;
    const int wave = tid >> 5;
    const int lane = tid & 31;

    // Stage X into LDS as bf16; zero-pad invalid slots.
    for (int i = tid; i < LMAX * DIM; i += 256) {
        int l = i / DIM, d = i % DIM;
        float v = (l < sz) ? emb[(size_t)(off + l) * DIM + d] : 0.0f;
        Xs[i] = (__bf16)v;
    }
    __syncthreads();

    v8f Sacc = {};  // waves 0..3 each own one 16x16 tile of S

    for (int ch = 0; ch < DIM / ECHUNK; ++ch) {
        // ---- XW chunk: M=32, N=128 -> 2x8 = 16 tiles over 8 waves ----
#pragma unroll
        for (int t = 0; t < 2; ++t) {
            int tile = wave * 2 + t;
            int mt = tile >> 3;            // 0..1
            int nt = tile & 7;             // 0..7
            int ecol = ch * ECHUNK + nt * 16;  // global W_sfa output row (e)
            v8f acc = {};
            for (int ks = 0; ks < DIM / 32; ++ks) {
                // prefetch next k-panel of the (L2-resident) weight row for this lane
                __builtin_prefetch(&Wsfa[(size_t)(ecol + (lane & 15)) * DIM +
                                         ((ks + 1) * 32 + (lane >> 4) * 16)], 0, 0);
                v16bf a = load_A_bf16(Xs, DIM, mt * 16, ks * 32, lane);
                v16bf b = load_Bt_f32(Wsfa, DIM, ecol, ks * 32, lane);
                acc = wmma_bf16(a, b, acc);
            }
            store_D_bf16(XWs, ECHUNK, mt * 16, nt * 16, lane, acc);
        }
        __syncthreads();

        // ---- S += X[:, chunk] @ XWchunk^T : 4 tiles on waves 0..3 ----
        if (wave < 4) {
            int mt = wave >> 1, nt = wave & 1;
            for (int ks = 0; ks < ECHUNK / 32; ++ks) {
                v16bf a = load_A_bf16(Xs, DIM, mt * 16, ch * ECHUNK + ks * 32, lane);
                v16bf b = load_Bt_bf16(XWs, ECHUNK, nt * 16, ks * 32, lane);
                Sacc = wmma_bf16(a, b, Sacc);
            }
        }
        __syncthreads();  // XWs reused next chunk
    }

    if (wave < 4) store_D_f32(Ss, LMAX, (wave >> 1) * 16, (wave & 1) * 16, lane, Sacc);
    __syncthreads();

    // max over valid l, then tanh (monotone: tanh(max) == max(tanh))
    if (tid < LMAX) {
        int m = tid;
        float mx = -3.0e38f;
        if (m < sz) {
            for (int l = 0; l < sz; ++l) mx = fmaxf(mx, Ss[l * LMAX + m]);
            mx = tanhf(mx);
        }
        att[m] = mx;
    }
    __syncthreads();

    if (tid == 0) {
        float mx = -3.0e38f;
        for (int m = 0; m < sz; ++m) mx = fmaxf(mx, att[m]);
        float s = 0.0f;
        for (int m = 0; m < sz; ++m) { float e = __expf(att[m] - mx); att[m] = e; s += e; }
        for (int m = sz; m < LMAX; ++m) att[m] = 0.0f;
        invsum = 1.0f / s;
    }
    __syncthreads();

    const float inv = invsum;
    for (int d = tid; d < DIM; d += 256) {
        float acc = 0.0f;
        for (int m = 0; m < sz; ++m) acc += att[m] * (float)Xs[m * DIM + d];
        pooled_out[(size_t)g * DIM + d] = (__bf16)(acc * inv);
    }
}

// ---------------------------------------------------------------------------
// Kernel 2: fused MLP score.  H = pooled @ W1^T, score[g] += relu(H+b1) . W2
// grid = (GRP/16, DHID/256); each block: 16 group-rows x 256 hidden cols.
//
// The 16x768 bf16 A-tile (24 KB) is shared by all 8 waves x 2 N-tiles, so it
// is staged into LDS once per block with the Tensor Data Mover:
//   tensor_load_to_lds D#(group0, group1)   -- 2-D tile, data_size=2B,
//   tile/tensor dim0 = 768 elems, dim1 = 16 rows, stride = 768.
// Wave 0 issues the DMA, waits on TENSORcnt, then a barrier publishes LDS.
// ---------------------------------------------------------------------------
__global__ __launch_bounds__(256) void mlp_kernel(const __bf16* __restrict__ pooled,
                                                  const float*  __restrict__ W1,
                                                  const float*  __restrict__ b1,
                                                  const float*  __restrict__ W2,
                                                  float*        __restrict__ scoreAcc) {
    __shared__ __align__(128) __bf16 Apool[16 * DIM];   // 24 KB staged A-tile
    __shared__ float part[16];

    const int tid  = threadIdx.x;
    const int wave = tid >> 5;
    const int lane = tid & 31;
    const int gbase = blockIdx.x * 16;
    const int jbase = blockIdx.y * 256;

    if (tid < 16) part[tid] = 0.0f;

    if (wave == 0) {
        // Build the Tensor DMA Descriptor (D#) groups 0/1 in SGPRs.
        unsigned long long gaddr =
            (unsigned long long)(size_t)(pooled + (size_t)gbase * DIM);
        unsigned int lds_off = (unsigned int)(size_t)&Apool[0];  // low 32b = LDS byte offset

        u32x4 g0;
        g0[0] = 1u;                                         // count=1, user-mode load
        g0[1] = lds_off;                                    // lds_addr (bytes)
        g0[2] = (unsigned int)(gaddr & 0xffffffffu);        // global_addr[31:0]
        g0[3] = (unsigned int)((gaddr >> 32) & 0x01ffffffu) // global_addr[56:32]
                | 0x80000000u;                              // type=2 ("image")

        u32x8 g1;
        g1[0] = 0x00010000u;   // workgroup_mask=0, data_size=1 (2 bytes)
        g1[1] = (unsigned int)(DIM) << 16;   // tensor_dim0[15:0]=768 in bits[63:48]
        g1[2] = 16u << 16;     // tensor_dim0 hi=0; tensor_dim1[15:0]=16 in bits[95:80]
        g1[3] = (unsigned int)(DIM) << 16;   // tensor_dim1 hi=0; tile_dim0=768
        g1[4] = 16u;           // tile_dim1=16, tile_dim2=0
        g1[5] = (unsigned int)(DIM);         // tensor_dim0_stride lo = 768
        g1[6] = 0u;            // stride hi=0, tensor_dim1_stride lo16=0
        g1[7] = 0u;

        asm volatile("tensor_load_to_lds %0, %1" :: "s"(g0), "s"(g1) : "memory");
        __builtin_amdgcn_s_wait_tensorcnt(0);
    }
    __syncthreads();   // A-tile visible to all waves

#pragma unroll
    for (int t = 0; t < 2; ++t) {
        int jcol = jbase + (wave * 2 + t) * 16;
        v8f acc = {};
        for (int ks = 0; ks < DIM / 32; ++ks) {
            v16bf a = load_A_bf16(Apool, DIM, 0, ks * 32, lane);
            v16bf b = load_Bt_f32(W1, DIM, jcol, ks * 32, lane);
            acc = wmma_bf16(a, b, acc);
        }
        int j   = jcol + (lane & 15);
        float bj = b1[j];
        float wj = W2[j];
        int m0 = (lane >> 4) * 8;   // local group-row base for this lane
#pragma unroll
        for (int r = 0; r < 8; ++r) {
            float h = acc[r] + bj;
            h = h > 0.0f ? h : 0.0f;
            atomicAdd(&part[m0 + r], h * wj);
        }
    }
    __syncthreads();
    if (tid < 16) atomicAdd(&scoreAcc[gbase + tid], part[tid]);
}

// ---------------------------------------------------------------------------
// Kernel 3: sigmoid + margin hinge loss reduction -> d_out[0]
// ---------------------------------------------------------------------------
__global__ __launch_bounds__(128) void loss_kernel(const float* __restrict__ scoreAcc,
                                                   const float* __restrict__ b2,
                                                   float* __restrict__ out) {
    __shared__ float partial[128];
    const int b = threadIdx.x;      // one thread per batch element
    const float bb = b2[0];
    float p = 1.0f / (1.0f + __expf(-(scoreAcc[b * (NEGS + 1)] + bb)));
    float acc = 0.0f;
    for (int j = 1; j <= NEGS; ++j) {
        float n = 1.0f / (1.0f + __expf(-(scoreAcc[b * (NEGS + 1) + j] + bb)));
        float v = n + GAMMA_F - p;
        acc += v > 0.0f ? v : 0.0f;
    }
    partial[b] = acc;
    __syncthreads();
    for (int s = 64; s > 0; s >>= 1) {
        if (b < s) partial[b] += partial[b + s];
        __syncthreads();
    }
    if (b == 0) out[0] = partial[0];
}

// ---------------------------------------------------------------------------
// Launcher. Inputs (setup_inputs order):
//   0 triple_emb f32 [N, 768]        1 W_sfa f32 [768, 768]
//   2 W1 f32 [3072, 768]             3 b1 f32 [3072]
//   4 W2 f32 [1, 3072]               5 b2 f32 [1]
//   6 tri2path_size i32 [128, 16]
// Output: f32 scalar loss.
// Workspace: offsets(int[2048]) | scoreAcc(f32[2048]) | pooled(bf16[2048*768])
// ---------------------------------------------------------------------------
extern "C" void kernel_launch(void* const* d_in, const int* in_sizes, int n_in,
                              void* d_out, int out_size, void* d_ws, size_t ws_size,
                              hipStream_t stream) {
    const float* emb   = (const float*)d_in[0];
    const float* Wsfa  = (const float*)d_in[1];
    const float* W1    = (const float*)d_in[2];
    const float* b1    = (const float*)d_in[3];
    const float* W2    = (const float*)d_in[4];
    const float* b2    = (const float*)d_in[5];
    const int*   sizes = (const int*)d_in[6];

    char* ws = (char*)d_ws;
    int*    offsets  = (int*)ws;                                  // 8 KB
    float*  scoreAcc = (float*)(ws + GRP * sizeof(int));          // 8 KB
    __bf16* pooled   = (__bf16*)(ws + GRP * sizeof(int) + GRP * sizeof(float));

    setup_kernel<<<1, 256, 0, stream>>>(sizes, offsets, scoreAcc);
    group_kernel<<<GRP, 256, 0, stream>>>(emb, Wsfa, offsets, sizes, pooled);
    mlp_kernel<<<dim3(GRP / 16, DHID / 256), 256, 0, stream>>>(pooled, W1, b1, W2, scoreAcc);
    loss_kernel<<<1, 128, 0, stream>>>(scoreAcc, b2, (float*)d_out);
}
